// VAE_49014166782517
// MI455X (gfx1250) — matrix operand order repaired and checked by reading, not compiled, not run on hardware
//
#include <hip/hip_runtime.h>
#include <hip/hip_bf16.h>
#include <math.h>

typedef __attribute__((ext_vector_type(16))) _Float16 v16h;
typedef __attribute__((ext_vector_type(8)))  _Float16 v8h;
typedef __attribute__((ext_vector_type(4)))  _Float16 v4h;
typedef __attribute__((ext_vector_type(8)))  float    v8f;
typedef __attribute__((ext_vector_type(4)))  float    v4f;

#define T_STEPS 512
#define BATCH   1024
#define HID     128
#define SP      136   // padded half-stride for LDS matrices (mult of 8 -> 16B aligned rows)
#define GP      516   // padded float-stride for gate buffer (mult of 4 -> 16B aligned rows)

__device__ __forceinline__ float fast_sigmoid(float x) {
    // v_exp + v_rcp; saturates correctly at +/-inf
    return __builtin_amdgcn_rcpf(1.f + __expf(-x));
}
__device__ __forceinline__ float fast_tanh(float x) {
    // tanh(x) = 1 - 2/(exp(2x)+1); v_exp + v_rcp
    return 1.f - 2.f * __builtin_amdgcn_rcpf(__expf(2.f * x) + 1.f);
}

// Build a 16x32 f16 A-fragment from an LDS tile (row stride SP halves), per the
// contiguous k order used here (same as B): lanes 0-15 hold row=lane K {k0..k0+15},
// lanes 16-31 hold row=lane-16 K {k0+16..k0+31}.
__device__ __forceinline__ v16h load_a_frag(const _Float16* base, int kb, int lane) {
    int row = lane & 15;
    int k0  = kb * 32 + ((lane & 16) ? 16 : 0);
    const _Float16* p = base + row * SP + k0;
    v8h lo = *(const v8h*)(p);
    v8h hi = *(const v8h*)(p + 8);
    return __builtin_shufflevector(lo, hi, 0,1,2,3,4,5,6,7,8,9,10,11,12,13,14,15);
}

// One LSTM layer. Each block handles 16 batch rows for all T steps.
// 512 threads = 16 waves; wave w owns gate-column tiles {2w, 2w+1} (N=16 each),
// holding its f16 weight B-fragments in VGPRs for the whole kernel.
template <int KXB, bool FIRST, bool WRITE_SEQ, bool WRITE_HT>
__global__ __launch_bounds__(512, 1)
void lstm_layer_kernel(const float* __restrict__ x0,      // [B, T, 32] (FIRST only)
                       _Float16*    __restrict__ seq,     // [T, B, 128] f16, in/out (in-place across layers)
                       const float* __restrict__ Wih,     // [512, KXB*32]
                       const float* __restrict__ Whh,     // [512, 128]
                       const float* __restrict__ bvec,    // [512]
                       float*       __restrict__ hT)      // [B, 128] (WRITE_HT only)
{
    constexpr int KX   = KXB * 32;
    constexpr int KBLK = KXB + 4;

    __shared__ __align__(16) _Float16 xs[16 * SP];
    __shared__ __align__(16) _Float16 hs[16 * SP];
    __shared__ __align__(16) float cs[16 * HID];
    __shared__ __align__(16) float gates[16 * GP];

    const int tid  = threadIdx.x;
    const int lane = tid & 31;
    const int wv   = tid >> 5;          // 0..15
    const int gb0  = blockIdx.x * 16;   // first global batch row of this tile

    // 4-wide element mapping used by all scalar phases: thread -> (erow, ej0)
    const int erow = tid >> 5;          // 0..15
    const int ej0  = (tid & 31) * 4;    // 0..124

    // ---- resident weight B-fragments (contiguous k order, same as A: lanes 0-15 = col n,
    // K k0..k0+15; lanes 16-31 = col n, K k0+16..k0+31; K contiguous per lane ->
    // contiguous along W's row since B[k,n] = W[col,k]).
    v16h wf[2][KBLK];
    float bias[2];
    const int khalf = (lane & 16) ? 16 : 0;
    #pragma unroll
    for (int i = 0; i < 2; ++i) {
        const int cb  = (2 * wv + i) * 16;
        const int col = cb + (lane & 15);
        bias[i] = bvec[col];
        #pragma unroll
        for (int kb = 0; kb < KBLK; ++kb) {
            const float* wr;
            int kst;
            if (kb < KXB) { wr = Wih + (size_t)col * KX;  kst = kb * 32 + khalf; }
            else          { wr = Whh + (size_t)col * HID; kst = (kb - KXB) * 32 + khalf; }
            v16h f;
            #pragma unroll
            for (int q = 0; q < 16; ++q) f[q] = (_Float16)wr[kst + q];
            wf[i][kb] = f;
        }
    }

    // ---- zero h, c (one vectorized write per thread)
    {
        v4h zh = { (_Float16)0.f, (_Float16)0.f, (_Float16)0.f, (_Float16)0.f };
        v4f zc = { 0.f, 0.f, 0.f, 0.f };
        *(v4h*)&hs[erow * SP + ej0] = zh;
        *(v4f*)&cs[erow * HID + ej0] = zc;
    }
    __syncthreads();

    #pragma unroll 1
    for (int t = 0; t < T_STEPS; ++t) {
        // ---- stage x_t into LDS (f16), vectorized 4-wide
        if (FIRST) {
            if (tid < 128) {
                int row = tid >> 3, k0 = (tid & 7) * 4;   // 16 rows x 32 cols
                v4f xv = *(const v4f*)&x0[(size_t)(gb0 + row) * T_STEPS * KX +
                                          (size_t)t * KX + k0];
                v4h xh;
                #pragma unroll
                for (int q = 0; q < 4; ++q) xh[q] = (_Float16)xv[q];
                *(v4h*)&xs[row * SP + k0] = xh;
            }
        } else {
            *(v4h*)&xs[erow * SP + ej0] =
                *(const v4h*)&seq[(size_t)t * BATCH * HID +
                                  (size_t)(gb0 + erow) * HID + ej0];
        }
        __syncthreads();

        // ---- build ALL A fragments for this step, then fence the scheduler so
        // the ds_load_b128s issue up front (no sinking/register-recycling that
        // would serialize load->wait->wmma).
        v16h afrag[KBLK];
        #pragma unroll
        for (int kb = 0; kb < KXB; ++kb) afrag[kb] = load_a_frag(xs, kb, lane);
        #pragma unroll
        for (int kb = 0; kb < 4; ++kb)   afrag[KXB + kb] = load_a_frag(hs, kb, lane);
        __builtin_amdgcn_sched_barrier(0);

        // ---- gate GEMM via WMMA: two interleaved accumulator chains (the two
        // N-tiles of this wave) so dependent-WMMA hazards are hidden.
        v8f acc0, acc1;
        #pragma unroll
        for (int r = 0; r < 8; ++r) { acc0[r] = bias[0]; acc1[r] = bias[1]; }
        #pragma unroll
        for (int kb = 0; kb < KBLK; ++kb) {
            acc0 = __builtin_amdgcn_wmma_f32_16x16x32_f16(
                    false, afrag[kb], false, wf[0][kb], (short)0, acc0, false, false);
            acc1 = __builtin_amdgcn_wmma_f32_16x16x32_f16(
                    false, afrag[kb], false, wf[1][kb], (short)0, acc1, false, false);
        }
        // C layout: lanes 0-15 -> rows r, lanes 16-31 -> rows r+8; col = cb + lane%16
        {
            const int col0 = (2 * wv) * 16 + (lane & 15);
            const int rb   = (lane & 16) ? 8 : 0;
            #pragma unroll
            for (int r = 0; r < 8; ++r) {
                gates[(rb + r) * GP + col0]      = acc0[r];
                gates[(rb + r) * GP + col0 + 16] = acc1[r];
            }
        }
        __syncthreads();

        // ---- elementwise LSTM cell update (torch gate order i,f,g,o), 4-wide
        {
            v4f gi = *(const v4f*)&gates[erow * GP + ej0];
            v4f gf = *(const v4f*)&gates[erow * GP + ej0 + 128];
            v4f gg = *(const v4f*)&gates[erow * GP + ej0 + 256];
            v4f go = *(const v4f*)&gates[erow * GP + ej0 + 384];
            v4f c  = *(const v4f*)&cs[erow * HID + ej0];
            v4h hv;
            #pragma unroll
            for (int q = 0; q < 4; ++q) {
                float i_ = fast_sigmoid(gi[q]);
                float f_ = fast_sigmoid(gf[q]);
                float g_ = fast_tanh(gg[q]);
                float o_ = fast_sigmoid(go[q]);
                float cn = f_ * c[q] + i_ * g_;
                c[q] = cn;
                hv[q] = (_Float16)(o_ * fast_tanh(cn));
            }
            *(v4f*)&cs[erow * HID + ej0] = c;
            *(v4h*)&hs[erow * SP + ej0] = hv;
            if (WRITE_SEQ)
                *(v4h*)&seq[(size_t)t * BATCH * HID +
                            (size_t)(gb0 + erow) * HID + ej0] = hv;
        }
        __syncthreads();
    }

    if (WRITE_HT) {
        v4h hv = *(const v4h*)&hs[erow * SP + ej0];
        v4f o;
        #pragma unroll
        for (int q = 0; q < 4; ++q) o[q] = (float)hv[q];
        *(v4f*)&hT[(size_t)(gb0 + erow) * HID + ej0] = o;
    }
}

// Fused VAE head: layernorm -> fc1 -> fc2 -> 4 heads -> reparam -> dec1 -> dec2 -> out.
// One block (128 threads) per batch row; total FLOPs negligible vs the LSTM.
__global__ __launch_bounds__(128)
void vae_head_kernel(const float* __restrict__ hT, const float* __restrict__ eps,
                     const float* __restrict__ u, const float* __restrict__ choice,
                     const float* __restrict__ ln_g, const float* __restrict__ ln_b,
                     const float* __restrict__ fc1_W, const float* __restrict__ fc1_b,
                     const float* __restrict__ fc2_W, const float* __restrict__ fc2_b,
                     const float* __restrict__ mean_W, const float* __restrict__ mean_b,
                     const float* __restrict__ logvar_W, const float* __restrict__ logvar_b,
                     const float* __restrict__ scale_W, const float* __restrict__ scale_b,
                     const float* __restrict__ shape_W, const float* __restrict__ shape_b,
                     const float* __restrict__ dec1_W, const float* __restrict__ dec1_b,
                     const float* __restrict__ dec2_W, const float* __restrict__ dec2_b,
                     const float* __restrict__ out_W, const float* __restrict__ out_b,
                     const float* __restrict__ pi_params, float* __restrict__ out)
{
    __shared__ float sh[128], s1[128], s2[64], sz[64], sd1[128], sd2[500];
    __shared__ float smu, srstd;
    const int b = blockIdx.x, tid = threadIdx.x;

    float h = hT[(size_t)b * 128 + tid];
    sh[tid] = h;
    __syncthreads();
    if (tid == 0) {
        float m = 0.f;
        for (int k = 0; k < 128; ++k) m += sh[k];
        m *= (1.f / 128.f);
        float v = 0.f;
        for (int k = 0; k < 128; ++k) { float d = sh[k] - m; v += d * d; }
        v *= (1.f / 128.f);
        smu = m; srstd = rsqrtf(v + 1e-5f);
    }
    __syncthreads();
    float hn = (h - smu) * srstd * ln_g[tid] + ln_b[tid];
    sh[tid] = hn;
    __syncthreads();
    {   // fc1: 128 -> 128
        float a = fc1_b[tid];
        const float* w = fc1_W + (size_t)tid * 128;
        for (int k = 0; k < 128; ++k) a += sh[k] * w[k];
        s1[tid] = fmaxf(a, 0.f);
    }
    __syncthreads();
    if (tid < 64) {  // fc2: 128 -> 64
        float a = fc2_b[tid];
        const float* w = fc2_W + (size_t)tid * 128;
        for (int k = 0; k < 128; ++k) a += s1[k] * w[k];
        s2[tid] = fmaxf(a, 0.f);
    }
    __syncthreads();
    if (tid < 64) {  // four heads + reparameterize
        float zm = mean_b[tid], lv = logvar_b[tid], sc = scale_b[tid], sp = shape_b[tid];
        const float* wm = mean_W  + (size_t)tid * 64;
        const float* wl = logvar_W + (size_t)tid * 64;
        const float* ws = scale_W + (size_t)tid * 64;
        const float* wp = shape_W + (size_t)tid * 64;
        for (int k = 0; k < 64; ++k) {
            float v = s2[k];
            zm += v * wm[k]; lv += v * wl[k]; sc += v * ws[k]; sp += v * wp[k];
        }
        sc = __expf(sc);
        sp = fmaxf(sp, 1e-6f);
        const size_t o = (size_t)b * 64 + tid;
        out[131072 + o] = zm;   // z_mean
        out[196608 + o] = lv;   // z_logvar
        out[262144 + o] = sc;   // z_scale
        out[327680 + o] = sp;   // z_shape
        float zg = zm + eps[o] * __expf(0.5f * lv);
        float uu = u[o];
        float zp = sc / sp * (__expf(-sp * __logf(1.f - uu)) - 1.f);
        float e0 = __expf(pi_params[0]), e1 = __expf(pi_params[1]);
        float pi0 = e0 / (e0 + e1);
        sz[tid] = (choice[b] < pi0) ? zg : zp;
    }
    __syncthreads();
    {   // dec1: 64 -> 128
        float a = dec1_b[tid];
        const float* w = dec1_W + (size_t)tid * 64;
        for (int k = 0; k < 64; ++k) a += sz[k] * w[k];
        sd1[tid] = fmaxf(a, 0.f);
    }
    __syncthreads();
    for (int j = tid; j < 500; j += 128) {  // dec2: 128 -> 500
        float a = dec2_b[j];
        const float* w = dec2_W + (size_t)j * 128;
        for (int k = 0; k < 128; ++k) a += sd1[k] * w[k];
        sd2[j] = fmaxf(a, 0.f);
    }
    __syncthreads();
    {   // out: 500 -> 128
        float a = out_b[tid];
        const float* w = out_W + (size_t)tid * 500;
        for (int k = 0; k < 500; ++k) a += sd2[k] * w[k];
        out[(size_t)b * 128 + tid] = a;
    }
}

extern "C" void kernel_launch(void* const* d_in, const int* in_sizes, int n_in,
                              void* d_out, int out_size, void* d_ws, size_t ws_size,
                              hipStream_t stream) {
    (void)in_sizes; (void)n_in; (void)out_size; (void)ws_size;
    const float* x      = (const float*)d_in[0];
    const float* eps    = (const float*)d_in[1];
    const float* u      = (const float*)d_in[2];
    const float* choice = (const float*)d_in[3];
    const float* Wih0 = (const float*)d_in[4],  *Whh0 = (const float*)d_in[5],  *b0 = (const float*)d_in[6];
    const float* Wih1 = (const float*)d_in[7],  *Whh1 = (const float*)d_in[8],  *b1 = (const float*)d_in[9];
    const float* Wih2 = (const float*)d_in[10], *Whh2 = (const float*)d_in[11], *b2 = (const float*)d_in[12];
    const float* ln_g = (const float*)d_in[13], *ln_b = (const float*)d_in[14];
    const float* fc1_W = (const float*)d_in[15], *fc1_b = (const float*)d_in[16];
    const float* fc2_W = (const float*)d_in[17], *fc2_b = (const float*)d_in[18];
    const float* mean_W = (const float*)d_in[19], *mean_b = (const float*)d_in[20];
    const float* logvar_W = (const float*)d_in[21], *logvar_b = (const float*)d_in[22];
    const float* scale_W = (const float*)d_in[23], *scale_b = (const float*)d_in[24];
    const float* shape_W = (const float*)d_in[25], *shape_b = (const float*)d_in[26];
    const float* dec1_W = (const float*)d_in[27], *dec1_b = (const float*)d_in[28];
    const float* dec2_W = (const float*)d_in[29], *dec2_b = (const float*)d_in[30];
    const float* out_W = (const float*)d_in[31], *out_b = (const float*)d_in[32];
    const float* pi_params = (const float*)d_in[33];
    float* out = (float*)d_out;

    // workspace: seq [T,B,128] f16 (reused in place across layers), then hT [B,128] f32
    _Float16* seq = (_Float16*)d_ws;
    float* hT = (float*)((char*)d_ws + (size_t)T_STEPS * BATCH * HID * sizeof(_Float16));

    const int blocks = BATCH / 16;
    lstm_layer_kernel<1, true,  true,  false><<<blocks, 512, 0, stream>>>(x, seq, Wih0, Whh0, b0, nullptr);
    lstm_layer_kernel<4, false, true,  false><<<blocks, 512, 0, stream>>>(nullptr, seq, Wih1, Whh1, b1, nullptr);
    lstm_layer_kernel<4, false, false, true ><<<blocks, 512, 0, stream>>>(nullptr, seq, Wih2, Whh2, b2, hT);

    vae_head_kernel<<<BATCH, 128, 0, stream>>>(hT, eps, u, choice, ln_g, ln_b,
        fc1_W, fc1_b, fc2_W, fc2_b, mean_W, mean_b, logvar_W, logvar_b,
        scale_W, scale_b, shape_W, shape_b, dec1_W, dec1_b, dec2_W, dec2_b,
        out_W, out_b, pi_params, out);
}
